// GRUDecoder_6004364280565
// MI455X (gfx1250) — matrix operand order repaired and checked by reading, not compiled, not run on hardware
//
#include <hip/hip_runtime.h>
#include <hip/hip_bf16.h>
#include <stdint.h>

// ---------------- problem constants ----------------
#define Bsz   64
#define Lsz   512
#define Esz   128
#define Hsz   512
#define ENCsz 256
#define Vsz   4
#define GVr   (Vsz + 2)   // embedding rows
#define SOS_TOK 4
#define NG    4           // batch groups (16 batches each)
#define WPG   16          // workgroups per group (== cluster size)
#define NT    512         // threads per workgroup (16 waves of 32)

typedef _Float16 h16;
typedef __attribute__((ext_vector_type(16))) _Float16 v16h;
typedef __attribute__((ext_vector_type(8)))  _Float16 v8h;
typedef __attribute__((ext_vector_type(8)))  float    v8f;

// ---------------- workspace layout (bytes) ----------------
#define GSTRIDE   262144
#define WS_ENC_H  ((size_t)0)                                   // f16 B*L*ENC
#define WS_KEYS   (WS_ENC_H + (size_t)Bsz*Lsz*ENCsz*2)          // f16 B*L*H
#define WS_WH     (WS_KEYS  + (size_t)Bsz*Lsz*Hsz*2)            // f16 H*H
#define WS_WS     (WS_WH    + (size_t)Hsz*Hsz*2)                // f16 H*ENC
#define WS_WIH0   (WS_WS    + (size_t)Hsz*ENCsz*2)              // f16 3H*(E+ENC)
#define WS_WHH0   (WS_WIH0  + (size_t)3*Hsz*(Esz+ENCsz)*2)      // f16 3H*H
#define WS_WIH1   (WS_WHH0  + (size_t)3*Hsz*Hsz*2)
#define WS_WHH1   (WS_WIH1  + (size_t)3*Hsz*Hsz*2)
#define WS_EMB    (WS_WHH1  + (size_t)3*Hsz*Hsz*2)              // f16 6*E
#define WS_GSTATE (WS_EMB   + (size_t)GVr*Esz*2 + 512)          // per-group state
#define WS_BARS   (WS_GSTATE + (size_t)NG*GSTRIDE)

// per-group state (double-buffered hidden states)
#define GS_H0G(b) ((b)*32768)                 // f32 16x512
#define GS_H1G(b) (65536 + (b)*32768)         // f32 16x512
#define GS_H0H(b) (131072 + (b)*16384)        // f16 16x512
#define GS_H1H(b) (163840 + (b)*16384)        // f16 16x512
#define GS_QH     196608                      // f16 16x512
#define GS_SC     212992                      // f32 16x512
#define GS_XG     245760                      // f16 16x384  [emb|ctx]

// ---------------- transcendentals: prefer native v_tanh_f32 ----------------
__device__ __forceinline__ float gpu_tanh(float x) {
#if __has_builtin(__builtin_amdgcn_tanhf)
  return __builtin_amdgcn_tanhf(x);
#else
  float e = __expf(2.0f * x);
  return 1.0f - 2.0f * __builtin_amdgcn_rcpf(e + 1.0f);
#endif
}
__device__ __forceinline__ float gpu_sig(float x) {
  return 0.5f + 0.5f * gpu_tanh(0.5f * x);
}

// ---------------- WMMA fragment builders (contiguous k = 16h + s, same order for A and B) ----------------
// A 16x32 f16: lanes 0-15 row M=lane hold K 0-15, lanes 16-31 hold K 16-31
__device__ __forceinline__ v16h frag_a(const h16* A, int lda, int kbase, int lane) {
  int m = lane & 15, ko = (lane >> 4) * 16;
  const h16* p = A + (size_t)m * lda + kbase;
  return *(const v16h*)(p + ko);
}
// B 32x16 f16 from row-major weight W (N x K): lane n = lane&15, K group = (lane>>4)*16
__device__ __forceinline__ v16h frag_b(const h16* W, int ldk, int n0, int kbase, int lane) {
  int n = lane & 15, kg = (lane >> 4) * 16;
  return *(const v16h*)(W + (size_t)(n0 + n) * ldk + kbase + kg);
}
__device__ __forceinline__ v8f wmma16(v16h a, v16h b, v8f c) {
  return __builtin_amdgcn_wmma_f32_16x16x32_f16(false, a, false, b, (short)0, c, false, false);
}

// ---------------- async global->LDS staging (ASYNCcnt path) ----------------
__device__ __forceinline__ void async_g2l_b128(uint32_t lds_off, const void* g) {
  asm volatile("global_load_async_to_lds_b128 %0, %1, off"
               :: "v"(lds_off), "v"((uint64_t)(uintptr_t)g) : "memory");
}
__device__ __forceinline__ void wait_async0() {
  asm volatile("s_wait_asynccnt 0" ::: "memory");
}

// ---------------- group barrier: cluster barrier + agent-scope atomic fallback ----
__device__ __forceinline__ void gsync(unsigned* bar) {
  __syncthreads();
  if (threadIdx.x == 0) {
    __builtin_amdgcn_s_cluster_barrier();   // HW path when cluster-dispatched (NOP otherwise)
    __threadfence();
    unsigned gen = __hip_atomic_load(bar + 1, __ATOMIC_ACQUIRE, __HIP_MEMORY_SCOPE_AGENT);
    unsigned n = __hip_atomic_fetch_add(bar, 1u, __ATOMIC_ACQ_REL, __HIP_MEMORY_SCOPE_AGENT);
    if (n == WPG - 1) {
      __hip_atomic_store(bar, 0u, __ATOMIC_RELAXED, __HIP_MEMORY_SCOPE_AGENT);
      __hip_atomic_fetch_add(bar + 1, 1u, __ATOMIC_RELEASE, __HIP_MEMORY_SCOPE_AGENT);
    } else {
      while (__hip_atomic_load(bar + 1, __ATOMIC_ACQUIRE, __HIP_MEMORY_SCOPE_AGENT) == gen)
        __builtin_amdgcn_s_sleep(2);
    }
  }
  __syncthreads();
}

struct Args {
  const float* enc; const int* tgt; const uint8_t* mask; const float* vvec;
  const float* bih0; const float* bhh0; const float* bih1; const float* bhh1;
  const float* Wout; const float* bout; const float* Winit; const float* binit;
  const h16 *enc_h, *Wh_h, *Ws_h, *Wih0_h, *Whh0_h, *Wih1_h, *Whh1_h, *emb_h;
  h16* keys;
  char* gbase;
  unsigned* bars;
  float* out;
};

// one GRU layer for this WG's h-slice [w*32, w*32+32): 12 waves each own one
// (n-tile, gate, gi/gh) accumulator; elementwise gate combine over LDS tiles.
__device__ __forceinline__ void gru_phase(
    const h16* Ain, int lda_in, int K_in, const h16* Wih, const float* bih,
    const h16* Ah, const h16* Whh, const float* bhh,
    const float* hold, float* hnew_f, h16* hnew_h,
    float* gtile, int w, int tid, int lane, int wv) {
  if (wv < 12) {
    int nt = wv / 6, part = wv % 6;
    int gate = part % 3;
    bool isH = part >= 3;
    const h16* Aa = isH ? Ah : Ain;
    int lda = isH ? Hsz : lda_in;
    int K = isH ? Hsz : K_in;
    int ldk = isH ? Hsz : K_in;
    const h16* Wb = (isH ? Whh : Wih) + (size_t)(gate * Hsz + w * 32 + nt * 16) * (size_t)ldk;
    v8f acc = {};
    for (int kc = 0; kc < K / 32; ++kc) {
      v16h a = frag_a(Aa, lda, kc * 32, lane);
      v16h bm = frag_b(Wb, ldk, 0, kc * 32, lane);
      acc = wmma16(a, bm, acc);
    }
    int c = lane & 15, rsel = (lane >> 4) * 8;
    float* dst = gtile + (nt * 6 + part) * 256;
#pragma unroll
    for (int r = 0; r < 8; ++r) dst[(rsel + r) * 16 + c] = acc[r];
  }
  __syncthreads();
  {
    int nt = tid >> 8, idx = tid & 255;
    int m = idx >> 4, c = idx & 15;
    int n = w * 32 + nt * 16 + c;
    const float* gt = gtile + nt * 6 * 256;
    float ir = gt[0 * 256 + idx] + bih[n];
    float iz = gt[1 * 256 + idx] + bih[Hsz + n];
    float ii = gt[2 * 256 + idx] + bih[2 * Hsz + n];
    float hr = gt[3 * 256 + idx] + bhh[n];
    float hz = gt[4 * 256 + idx] + bhh[Hsz + n];
    float hn = gt[5 * 256 + idx] + bhh[2 * Hsz + n];
    float rg = gpu_sig(ir + hr);
    float zg = gpu_sig(iz + hz);
    float ng = gpu_tanh(ii + rg * hn);
    float ho = hold[(size_t)m * Hsz + n];
    float h = (1.f - zg) * ng + zg * ho;
    hnew_f[(size_t)m * Hsz + n] = h;
    hnew_h[(size_t)m * Hsz + n] = (h16)h;
  }
  __syncthreads();
}

__global__ void k_cvt(const float* s, h16* d, int n) {
  int i = blockIdx.x * blockDim.x + threadIdx.x;
  if (i < n) d[i] = (h16)s[i];
}
__global__ void k_zero(unsigned* p, int n) {
  int i = blockIdx.x * blockDim.x + threadIdx.x;
  if (i < n) p[i] = 0u;
}

__global__ __launch_bounds__(NT, 1)
void decode_kernel(Args A) {
  const int g = blockIdx.x / WPG, w = blockIdx.x % WPG;
  const int b0 = g * 16;
  const int tid = threadIdx.x, lane = tid & 31, wv = tid >> 5;
  char* G = A.gbase + (size_t)g * GSTRIDE;
  unsigned* bar = A.bars + g * 2;

  __shared__ h16 qs[16 * Hsz];          // 16KB staged q
  __shared__ float srow[Lsz];
  __shared__ float red[32];
  __shared__ float ctx2[2 * ENCsz];
  __shared__ float gtile[2 * 6 * 256];  // 12KB gate tiles

  // ---- init hidden for this WG's row r=w: tanh(meanpool @ Winit^T + binit)
  {
    const int bg = b0 + w;
    if (tid < ENCsz) {
      float s = 0.f; int c = 0;
      for (int l = 0; l < Lsz; ++l) {
        if (A.mask[(size_t)bg * Lsz + l]) { s += A.enc[((size_t)bg * Lsz + l) * ENCsz + tid]; c++; }
      }
      srow[tid] = s / (float)(c > 0 ? c : 1);
    }
    __syncthreads();
    float* h0g = (float*)(G + GS_H0G(0));
    float* h1g = (float*)(G + GS_H1G(0));
    h16* h0h = (h16*)(G + GS_H0H(0));
    h16* h1h = (h16*)(G + GS_H1H(0));
    for (int rep = 0; rep < 2; ++rep) {
      int j = tid + rep * NT;
      float acc = A.binit[j];
      const float* Wr = A.Winit + (size_t)j * ENCsz;
      for (int e = 0; e < ENCsz; ++e) acc += srow[e] * Wr[e];
      float h = gpu_tanh(acc);
      if (j < Hsz) { h0g[w * Hsz + j] = h; h0h[w * Hsz + j] = (h16)h; }
      else         { h1g[w * Hsz + (j - Hsz)] = h; h1h[w * Hsz + (j - Hsz)] = (h16)h; }
    }
    __syncthreads();
  }

  // ---- keys precompute (WMMA, M = l): keys[b,l,:] = enc[b,l,:] @ Ws^T
  {
    const int l0 = w * 32;
    for (int tI = wv; tI < 1024; tI += 16) {
      int b = tI >> 6, mt = (tI >> 5) & 1, nt = tI & 31;
      const h16* Arow = A.enc_h + ((size_t)(b0 + b) * Lsz + l0 + mt * 16) * ENCsz;
      v8f acc = {};
#pragma unroll
      for (int kc = 0; kc < ENCsz / 32; ++kc) {
        v16h a = frag_a(Arow, ENCsz, kc * 32, lane);
        v16h bm = frag_b(A.Ws_h, ENCsz, nt * 16, kc * 32, lane);
        acc = wmma16(a, bm, acc);
      }
      int col = nt * 16 + (lane & 15);
      int rbase = l0 + mt * 16 + (lane >> 4) * 8;
      h16* kp = A.keys + (size_t)(b0 + b) * Lsz * Hsz;
#pragma unroll
      for (int r = 0; r < 8; ++r) kp[(size_t)(rbase + r) * Hsz + col] = (h16)acc[r];
    }
  }
  gsync(bar);

  for (int t = 0; t < Lsz; ++t) {
    const int rd = t & 1, wr = 1 - rd;
    const float* h0g_r = (const float*)(G + GS_H0G(rd));
    const float* h1g_r = (const float*)(G + GS_H1G(rd));
    float* h0g_w = (float*)(G + GS_H0G(wr));
    float* h1g_w = (float*)(G + GS_H1G(wr));
    const h16* h0h_r = (const h16*)(G + GS_H0H(rd));
    const h16* h1h_r = (const h16*)(G + GS_H1H(rd));
    h16* h0h_w = (h16*)(G + GS_H0H(wr));
    h16* h1h_w = (h16*)(G + GS_H1H(wr));
    h16* qh = (h16*)(G + GS_QH);
    float* sc = (float*)(G + GS_SC);
    h16* xg = (h16*)(G + GS_XG);

    // ---- P0: q slice (WMMA, M=16 batch rows) + logits of step t-1
    if (wv < 2) {
      int nt = w * 2 + wv;
      v8f acc = {};
#pragma unroll
      for (int kc = 0; kc < Hsz / 32; ++kc) {
        v16h a = frag_a(h1h_r, Hsz, kc * 32, lane);
        v16h bm = frag_b(A.Wh_h, Hsz, nt * 16, kc * 32, lane);
        acc = wmma16(a, bm, acc);
      }
      int col = nt * 16 + (lane & 15);
      int rsel = (lane >> 4) * 8;
#pragma unroll
      for (int r = 0; r < 8; ++r) qh[(size_t)(rsel + r) * Hsz + col] = (h16)acc[r];
    }
    if (w == 0 && t > 0 && tid < 16 * Vsz) {
      int r = tid >> 2, vv = tid & 3;
      const float* hr = h1g_r + r * Hsz;
      const float* wo = A.Wout + vv * Hsz;
      float acc = A.bout[vv];
      for (int k = 0; k < Hsz; ++k) acc += hr[k] * wo[k];
      A.out[(((size_t)(b0 + r)) * Lsz + (t - 1)) * Vsz + vv] = acc;
    }
    gsync(bar);

    // ---- P1: async-stage q into LDS, then attention energies (tanh-bound).
    // Each lane owns a contiguous 16-h chunk: q and v hoisted into registers,
    // keys read as 2x b128 per l, native v_tanh in the inner loop.
    {
      uint32_t loff = (uint32_t)(uintptr_t)qs + tid * 32;
      const char* src = (const char*)qh + tid * 32;
      async_g2l_b128(loff, src);
      async_g2l_b128(loff + 16, src + 16);
      wait_async0();
      __syncthreads();

      const int l0 = w * 32;
      const int bg = b0 + wv;   // one wave per batch row
      float vf[16], qf[16];
      {
        const float* vp = A.vvec + lane * 16;
        const v8h* qp = (const v8h*)(qs + wv * Hsz + lane * 16);
        v8h q0 = qp[0], q1 = qp[1];
#pragma unroll
        for (int j = 0; j < 8; ++j) {
          vf[j] = vp[j]; vf[8 + j] = vp[8 + j];
          qf[j] = (float)q0[j]; qf[8 + j] = (float)q1[j];
        }
      }
      const h16* kbase = A.keys + ((size_t)bg * Lsz + l0) * Hsz + lane * 16;
      for (int i = 0; i < 32; ++i) {
        const v8h* kp = (const v8h*)(kbase + (size_t)i * Hsz);
        v8h k0 = kp[0], k1 = kp[1];
        float s = 0.f;
#pragma unroll
        for (int j = 0; j < 8; ++j) {
          s += vf[j]     * gpu_tanh(qf[j]     + (float)k0[j]);
          s += vf[8 + j] * gpu_tanh(qf[8 + j] + (float)k1[j]);
        }
#pragma unroll
        for (int o = 16; o > 0; o >>= 1) s += __shfl_xor(s, o, 32);
        if (lane == 0) {
          bool mk = A.mask[(size_t)bg * Lsz + l0 + i] != 0;
          sc[wv * Lsz + l0 + i] = mk ? s : -3.0e38f;
        }
      }
    }
    gsync(bar);

    // ---- P2: softmax + context + x row (one batch row per WG)
    {
      const int r = w, bg = b0 + w;
      float s = sc[r * Lsz + tid];
      float m = s;
#pragma unroll
      for (int o = 16; o > 0; o >>= 1) m = fmaxf(m, __shfl_xor(m, o, 32));
      if (lane == 0) red[wv] = m;
      __syncthreads();
      if (tid == 0) { float x = red[0]; for (int i = 1; i < 16; ++i) x = fmaxf(x, red[i]); red[16] = x; }
      __syncthreads();
      float e = __expf(s - red[16]);
      float ss = e;
#pragma unroll
      for (int o = 16; o > 0; o >>= 1) ss += __shfl_xor(ss, o, 32);
      if (lane == 0) red[wv] = ss;
      __syncthreads();
      if (tid == 0) { float x = 0.f; for (int i = 0; i < 16; ++i) x += red[i]; red[17] = __builtin_amdgcn_rcpf(x); }
      __syncthreads();
      srow[tid] = e * red[17];
      __syncthreads();
      {
        int ee = tid & 255, seg = tid >> 8;
        float acc = 0.f;
        const h16* eb = A.enc_h + ((size_t)bg * Lsz + seg * 256) * ENCsz + ee;
        for (int l = 0; l < 256; ++l) acc += srow[seg * 256 + l] * (float)eb[(size_t)l * ENCsz];
        ctx2[seg * ENCsz + ee] = acc;
      }
      __syncthreads();
      if (tid < ENCsz) xg[r * 384 + Esz + tid] = (h16)(ctx2[tid] + ctx2[ENCsz + tid]);
      if (tid < Esz) {
        int tok = (t == 0) ? SOS_TOK : A.tgt[(size_t)bg * Lsz + (t - 1)];
        xg[r * 384 + tid] = A.emb_h[(size_t)tok * Esz + tid];
      }
    }
    gsync(bar);

    // ---- P3: GRU layer 0 ----
    gru_phase(xg, 384, Esz + ENCsz, A.Wih0_h, A.bih0,
              h0h_r, A.Whh0_h, A.bhh0, h0g_r, h0g_w, h0h_w,
              gtile, w, tid, lane, wv);
    gsync(bar);

    // ---- P4: GRU layer 1 ----
    gru_phase(h0h_w, Hsz, Hsz, A.Wih1_h, A.bih1,
              h1h_r, A.Whh1_h, A.bhh1, h1g_r, h1g_w, h1h_w,
              gtile, w, tid, lane, wv);
    gsync(bar);
  }

  // ---- final logits (t = L-1) ----
  if (w == 0 && tid < 16 * Vsz) {
    const float* h1f = (const float*)(G + GS_H1G(Lsz & 1));
    int r = tid >> 2, vv = tid & 3;
    float acc = A.bout[vv];
    const float* hr = h1f + r * Hsz;
    const float* wo = A.Wout + vv * Hsz;
    for (int k = 0; k < Hsz; ++k) acc += hr[k] * wo[k];
    A.out[(((size_t)(b0 + r)) * Lsz + (Lsz - 1)) * Vsz + vv] = acc;
  }
}

extern "C" void kernel_launch(void* const* d_in, const int* in_sizes, int n_in,
                              void* d_out, int out_size, void* d_ws, size_t ws_size,
                              hipStream_t stream) {
  char* ws = (char*)d_ws;
  const float* enc = (const float*)d_in[0];
  const int* tgt = (const int*)d_in[1];
  const uint8_t* mask = (const uint8_t*)d_in[2];
  const float* emb = (const float*)d_in[4];
  const float* Wh = (const float*)d_in[5];
  const float* Wsm = (const float*)d_in[6];
  const float* vvec = (const float*)d_in[7];
  const float* Wih0 = (const float*)d_in[8];
  const float* Whh0 = (const float*)d_in[9];
  const float* bih0 = (const float*)d_in[10];
  const float* bhh0 = (const float*)d_in[11];
  const float* Wih1 = (const float*)d_in[12];
  const float* Whh1 = (const float*)d_in[13];
  const float* bih1 = (const float*)d_in[14];
  const float* bhh1 = (const float*)d_in[15];
  const float* Wout = (const float*)d_in[16];
  const float* bout = (const float*)d_in[17];
  const float* Winit = (const float*)d_in[18];
  const float* binit = (const float*)d_in[19];

  auto cvt = [&](const float* s, size_t off, int n) {
    k_cvt<<<(n + 255) / 256, 256, 0, stream>>>(s, (h16*)(ws + off), n);
  };
  cvt(enc, WS_ENC_H, Bsz * Lsz * ENCsz);
  cvt(Wh, WS_WH, Hsz * Hsz);
  cvt(Wsm, WS_WS, Hsz * ENCsz);
  cvt(Wih0, WS_WIH0, 3 * Hsz * (Esz + ENCsz));
  cvt(Whh0, WS_WHH0, 3 * Hsz * Hsz);
  cvt(Wih1, WS_WIH1, 3 * Hsz * Hsz);
  cvt(Whh1, WS_WHH1, 3 * Hsz * Hsz);
  cvt(emb, WS_EMB, GVr * Esz);
  k_zero<<<1, 32, 0, stream>>>((unsigned*)(ws + WS_BARS), 8);

  Args A;
  A.enc = enc; A.tgt = tgt; A.mask = mask; A.vvec = vvec;
  A.bih0 = bih0; A.bhh0 = bhh0; A.bih1 = bih1; A.bhh1 = bhh1;
  A.Wout = Wout; A.bout = bout; A.Winit = Winit; A.binit = binit;
  A.enc_h = (const h16*)(ws + WS_ENC_H);
  A.Wh_h = (const h16*)(ws + WS_WH);
  A.Ws_h = (const h16*)(ws + WS_WS);
  A.Wih0_h = (const h16*)(ws + WS_WIH0);
  A.Whh0_h = (const h16*)(ws + WS_WHH0);
  A.Wih1_h = (const h16*)(ws + WS_WIH1);
  A.Whh1_h = (const h16*)(ws + WS_WHH1);
  A.emb_h = (const h16*)(ws + WS_EMB);
  A.keys = (h16*)(ws + WS_KEYS);
  A.gbase = ws + WS_GSTATE;
  A.bars = (unsigned*)(ws + WS_BARS);
  A.out = (float*)d_out;

  decode_kernel<<<NG * WPG, NT, 0, stream>>>(A);
  (void)in_sizes; (void)n_in; (void)out_size; (void)ws_size;
}
